// LlamaAttention_heavy_hitter_90134183674297
// MI455X (gfx1250) — matrix operand order repaired and checked by reading, not compiled, not run on hardware
//
#include <hip/hip_runtime.h>
#include <hip/hip_bf16.h>
#include <math.h>

// Problem constants (match reference)
#define S_LEN   2048
#define HID     4096
#define NH      32
#define NKV     8
#define HD      128
#define INIT_W  128
#define RECENT_W 512
#define HEAVY_K 512
#define QK_SCALE 0.08838834764831845f  // 1/sqrt(128)

typedef __attribute__((ext_vector_type(16))) __bf16 v16bf;
typedef __attribute__((ext_vector_type(8)))  float  v8f;
typedef __attribute__((ext_vector_type(4))) unsigned tdm_g0_t;
typedef __attribute__((ext_vector_type(8))) int      tdm_g1_t;
typedef __attribute__((ext_vector_type(4))) int      tdm_g2_t;

// ---------------------------------------------------------------------------
// helpers
// ---------------------------------------------------------------------------
__device__ __forceinline__ unsigned f2ord(float f) {
  // order-preserving float->uint map (descending compare == uint compare)
  unsigned u = __float_as_uint(f);
  return u ^ ((u & 0x80000000u) ? 0xFFFFFFFFu : 0x80000000u);
}

__device__ __forceinline__ float wave_max32(float v) {
  #pragma unroll
  for (int o = 16; o > 0; o >>= 1) v = fmaxf(v, __shfl_xor(v, o, 32));
  return v;
}
__device__ __forceinline__ float wave_sum32(float v) {
  #pragma unroll
  for (int o = 16; o > 0; o >>= 1) v += __shfl_xor(v, o, 32);
  return v;
}

// ---------------------------------------------------------------------------
// fp32 -> bf16 elementwise
// ---------------------------------------------------------------------------
__global__ void hh_f32_to_bf16(const float* __restrict__ src,
                               __bf16* __restrict__ dst, size_t n) {
  size_t i = (size_t)blockIdx.x * blockDim.x + threadIdx.x;
  size_t stride = (size_t)gridDim.x * blockDim.x;
  for (; i < n; i += stride) dst[i] = (__bf16)src[i];
}

// fp32 [K][N] -> bf16 transposed [N][K] (weights become K-contiguous so WMMA
// B-fragments are two contiguous global_load_b128 straight from L2)
__global__ void hh_f32_to_bf16_T(const float* __restrict__ src,
                                 __bf16* __restrict__ dst, int K, int N) {
  size_t total = (size_t)K * N;
  size_t i = (size_t)blockIdx.x * blockDim.x + threadIdx.x;
  size_t stride = (size_t)gridDim.x * blockDim.x;
  for (; i < total; i += stride) {
    int n = (int)(i / K);
    int k = (int)(i % K);
    dst[i] = (__bf16)src[(size_t)k * N + n];
  }
}

// ---------------------------------------------------------------------------
// bf16 GEMM: C[M,N] (f32) = A[M,K] x BT[N,K] (B pre-transposed, K-contiguous).
// Block tile 64x64, 8 waves; wave (wid&3, wid>>2) computes 16x32.
// A tile staged in LDS via GLOBAL_LOAD_ASYNC_TO_LDS_B128 (ASYNCcnt path);
// B fragments read directly from L2-resident transposed weights.
// ---------------------------------------------------------------------------
__global__ __launch_bounds__(256)
void hh_gemm_bf16(const __bf16* __restrict__ A, const __bf16* __restrict__ BT,
                  float* __restrict__ C, int M, int N, int K) {
  __shared__ __bf16 As[64 * 32];   // [m][k]

  const int bm = blockIdx.y * 64;
  const int bn = blockIdx.x * 64;
  const int tid  = threadIdx.x;
  const int wid  = tid >> 5;
  const int lane = tid & 31;
  const int wm = (wid & 3) * 16;     // wave M offset in block
  const int wn = (wid >> 2) * 32;    // wave N offset in block (2 frags)
  const int qrow  = lane & 15;
  const int khalf = lane >> 4;

  // per-thread A-staging addresses (16 B per thread per K-step)
  const int aidx = tid * 8;
  const int ar = aidx >> 5, ac = aidx & 31;
  const unsigned lds_dst = (unsigned)(size_t)(&As[ar * 32 + ac]);

  v8f acc0 = {}, acc1 = {};

  for (int k0 = 0; k0 < K; k0 += 32) {
    __syncthreads();   // previous iteration's fragment reads are done
    {
      const __bf16* src = A + (size_t)(bm + ar) * K + k0 + ac;
      unsigned long long ga = (unsigned long long)(size_t)src;
      // async DMA: global -> LDS, 16 bytes per lane, tracked by ASYNCcnt
      asm volatile("global_load_async_to_lds_b128 %0, %1, off"
                   :: "v"(lds_dst), "v"(ga) : "memory");
    }
    asm volatile("s_wait_asynccnt 0x0" ::: "memory");
    __syncthreads();

    // A fragment: row M=qrow; lanes<16 hold K{0..15}, lanes>=16 K{16..31}
    v16bf a;
    const __bf16* ap = As + (wm + qrow) * 32;
    #pragma unroll
    for (int i = 0; i < 8; ++i) { a[i] = ap[khalf * 16 + i]; a[8 + i] = ap[khalf * 16 + 8 + i]; }

    // B fragments from transposed weights: col N fixed -> 16 contiguous K elems
    v16bf b0, b1;
    const __bf16* bp0 = BT + (size_t)(bn + wn + qrow) * K + k0 + khalf * 16;
    const __bf16* bp1 = BT + (size_t)(bn + wn + 16 + qrow) * K + k0 + khalf * 16;
    #pragma unroll
    for (int i = 0; i < 16; ++i) { b0[i] = bp0[i]; b1[i] = bp1[i]; }

    acc0 = __builtin_amdgcn_wmma_f32_16x16x32_bf16(false, a, false, b0, (short)0, acc0, false, false);
    acc1 = __builtin_amdgcn_wmma_f32_16x16x32_bf16(false, a, false, b1, (short)0, acc1, false, false);
  }

  // C/D layout: element i -> M = khalf*8+i, N = qrow
  #pragma unroll
  for (int i = 0; i < 8; ++i) {
    size_t r = (size_t)(bm + wm + khalf * 8 + i);
    C[r * N + bn + wn + qrow]      = acc0[i];
    C[r * N + bn + wn + 16 + qrow] = acc1[i];
  }
}

// ---------------------------------------------------------------------------
// RoPE + scale + convert + head-major repack.
// src: f32 [S][Hn*HD]; dst: bf16 [Hn][S][HD]; grid = Hn*S blocks of HD threads.
// ---------------------------------------------------------------------------
__global__ __launch_bounds__(128)
void hh_rope_pack(const float* __restrict__ src, __bf16* __restrict__ dst,
                  int Hn, float scale) {
  const int h = blockIdx.x / S_LEN;
  const int s = blockIdx.x % S_LEN;
  const int d = threadIdx.x;
  const int j = d & 63;
  const float inv = __powf(10000.0f, -(float)j / 64.0f);
  float sn, cs;
  __sincosf((float)s * inv, &sn, &cs);
  const size_t row = (size_t)s * (Hn * HD) + h * HD;
  const float x  = src[row + d];
  const float xo = (d < 64) ? src[row + d + 64] : src[row + d - 64];
  const float r  = (d < 64) ? (x * cs - xo * sn) : (x * cs + xo * sn);
  dst[((size_t)h * S_LEN + s) * HD + d] = (__bf16)(r * scale);
}

// V: f32 [S][NKV*HD] -> bf16 transposed [NKV][HD][S] (so PV B-frags are contiguous)
__global__ void hh_v_packT(const float* __restrict__ src, __bf16* __restrict__ dst) {
  size_t total = (size_t)NKV * HD * S_LEN;
  size_t i = (size_t)blockIdx.x * blockDim.x + threadIdx.x;
  size_t stride = (size_t)gridDim.x * blockDim.x;
  for (; i < total; i += stride) {
    int kvh = (int)(i / ((size_t)HD * S_LEN));
    int rem = (int)(i % ((size_t)HD * S_LEN));
    int d = rem / S_LEN, s = rem % S_LEN;
    dst[i] = (__bf16)src[(size_t)s * (NKV * HD) + kvh * HD + d];
  }
}

// ---------------------------------------------------------------------------
// Exact k-th-largest threshold over srow[mlo..mhi] via 4-pass radix select
// on order-preserving bits. One wave; hist/scr are wave-private LDS.
// ---------------------------------------------------------------------------
__device__ unsigned hh_radix_thresh(const float* __restrict__ srow, int mlo, int mhi,
                                    unsigned* hist, unsigned* scr, int lane) {
  unsigned prefix = 0, pmask = 0;
  int k = HEAVY_K;
  #pragma unroll 1
  for (int pass = 0; pass < 4; ++pass) {
    const int shift = 24 - 8 * pass;
    for (int b = lane; b < 256; b += 32) hist[b] = 0u;
    __asm__ volatile("s_wait_dscnt 0x0" ::: "memory");
    for (int i = mlo + lane; i <= mhi; i += 32) {
      unsigned u = f2ord(srow[i]);
      if ((u & pmask) == prefix) atomicAdd(&hist[(u >> shift) & 255u], 1u);
    }
    __asm__ volatile("s_wait_dscnt 0x0" ::: "memory");
    if (lane == 0) {
      unsigned cum = 0, bin = 0, kk = (unsigned)k;
      for (int b = 255; b >= 0; --b) {
        unsigned c = hist[b];
        if (cum + c >= kk) { bin = (unsigned)b; kk -= cum; break; }
        cum += c;
      }
      scr[0] = bin; scr[1] = kk;
    }
    __asm__ volatile("s_wait_dscnt 0x0" ::: "memory");
    prefix |= scr[0] << shift;
    pmask  |= 0xFFu << shift;
    k = (int)scr[1];
  }
  return prefix;
}

// ---------------------------------------------------------------------------
// Attention: one block per (head, 16-query tile). 8 waves.
// Phase 0: TDM tensor_load_to_lds stages the Q tile (16x128 bf16) into LDS,
//          overlapped with the score-LDS memset; s_wait_tensorcnt + barrier.
// Phase 1: scores(16 x S) = Q_tile x K^T via WMMA into LDS (q pre-scaled).
// Phase 2: per-row exact top-512 threshold on "middle" band + masked softmax.
// Phase 3: out(16 x 128) = P x V via WMMA, 1/sum folded into store.
// ---------------------------------------------------------------------------
__global__ __launch_bounds__(256)
void hh_attn_kernel(const __bf16* __restrict__ qbf,   // [NH][S][HD]
                    const __bf16* __restrict__ kbf,   // [NKV][S][HD]
                    const __bf16* __restrict__ vbfT,  // [NKV][HD][S]
                    __bf16* __restrict__ attn_bf) {   // [S][NH*HD]
  extern __shared__ float smem[];
  float*    sc     = smem;                                   // [16][S]
  unsigned* hist   = (unsigned*)(sc + 16 * S_LEN);           // [8][256]
  unsigned* scr    = hist + 8 * 256;                         // [8][2]
  float*    rowinv = (float*)(scr + 16);                     // [16]
  __bf16*   qtile  = (__bf16*)(rowinv + 16);                 // [16][HD]

  const int tile = blockIdx.x % (S_LEN / 16);
  const int h    = blockIdx.x / (S_LEN / 16);
  const int kvh  = h / (NH / NKV);
  const int tid  = threadIdx.x;
  const int wid  = tid >> 5;
  const int lane = tid & 31;
  const int qrow  = lane & 15;
  const int khalf = lane >> 4;

  // ---- Phase 0: TDM DMA of the Q tile (issued by wave 0, overlaps memset) ----
  if (wid == 0) {
    const __bf16* gsrc = qbf + (((size_t)h * S_LEN) + tile * 16) * HD;
    unsigned long long ga = (unsigned long long)(size_t)gsrc;
    tdm_g0_t g0;
    g0[0] = 1u;                                            // count=1 (valid user D#)
    g0[1] = (unsigned)(size_t)qtile;                       // lds_addr
    g0[2] = (unsigned)(ga & 0xFFFFFFFFu);                  // global_addr[31:0]
    g0[3] = (unsigned)((ga >> 32) & 0x01FFFFFFu)           // global_addr[56:32]
          | (2u << 30);                                    // type=2 ("image")
    tdm_g1_t g1;
    g1[0] = (int)(1u << 16);       // workgroup_mask=0 (not in cluster), data_size=1 (2B)
    g1[1] = (int)(128u << 16);     // tensor_dim0[15:0]=HD=128 (in [31:16])
    g1[2] = (int)(16u << 16);      // tensor_dim0[31:16]=0 ; tensor_dim1[15:0]=16
    g1[3] = (int)(128u << 16);     // tensor_dim1[31:16]=0 ; tile_dim0=128
    g1[4] = 16;                    // tile_dim1=16 ; tile_dim2=0
    g1[5] = 128;                   // tensor_dim0_stride[31:0]=128 elements
    g1[6] = 0;                     // stride hi / dim1_stride lo
    g1[7] = 0;
    tdm_g2_t g2 = {0, 0, 0, 0};    // 2D tensor: groups 2/3 unused
    tdm_g2_t g3 = {0, 0, 0, 0};
    tdm_g1_t g4 = {0, 0, 0, 0, 0, 0, 0, 0};   // extra group (6-arg toolchain form)
    __builtin_amdgcn_tensor_load_to_lds(g0, g1, g2, g3, g4, 0);
  }

  for (int i = tid; i < 16 * S_LEN; i += 256) sc[i] = 0.0f;

  if (wid == 0) __builtin_amdgcn_s_wait_tensorcnt(0);
  __syncthreads();

  // ---- Phase 1: scores ----
  v16bf afr[4];   // Q A-fragments for the 4 d-chunks of 32, from LDS qtile
  {
    const __bf16* qp = qtile + qrow * HD;
    #pragma unroll
    for (int dk = 0; dk < 4; ++dk) {
      const __bf16* p0 = qp + dk * 32 + khalf * 16;
      const __bf16* p1 = qp + dk * 32 + khalf * 16 + 8;
      #pragma unroll
      for (int i = 0; i < 8; ++i) { afr[dk][i] = p0[i]; afr[dk][8 + i] = p1[i]; }
    }
  }
  for (int kf = wid; kf <= tile; kf += 8) {   // causal: skip key tiles beyond query tile
    v8f acc = {};
    #pragma unroll
    for (int dk = 0; dk < 4; ++dk) {
      const __bf16* bp = kbf + (((size_t)kvh * S_LEN) + kf * 16 + qrow) * HD
                             + dk * 32 + khalf * 16;
      v16bf bfr;
      #pragma unroll
      for (int i = 0; i < 16; ++i) bfr[i] = bp[i];
      acc = __builtin_amdgcn_wmma_f32_16x16x32_bf16(false, afr[dk], false, bfr,
                                                    (short)0, acc, false, false);
    }
    #pragma unroll
    for (int i = 0; i < 8; ++i)
      sc[(khalf * 8 + i) * S_LEN + kf * 16 + qrow] = acc[i];
  }
  __syncthreads();

  // ---- Phase 2: per-row heavy-hitter select + softmax (2 rows per wave) ----
  for (int rr = 0; rr < 2; ++rr) {
    const int row = wid * 2 + rr;
    const int qi  = tile * 16 + row;
    float* srow = sc + row * S_LEN;
    const int mlo = INIT_W, mhi = qi - RECENT_W;

    unsigned thr = 0u;   // 0 => keep entire middle band (<=512 candidates)
    if (mhi - mlo + 1 > HEAVY_K)
      thr = hh_radix_thresh(srow, mlo, mhi, hist + wid * 256, scr + wid * 2, lane);

    float m = -3.0e38f;
    for (int i = lane; i <= qi; i += 32) {
      const bool mid = (i >= mlo) && (i <= mhi);
      const float s = srow[i];
      if (!mid || (f2ord(s) >= thr)) m = fmaxf(m, s);
    }
    m = wave_max32(m);

    float sum = 0.0f;
    for (int i = lane; i < S_LEN; i += 32) {
      const float s = srow[i];
      const bool mid  = (i >= mlo) && (i <= mhi);
      const bool keep = (i <= qi) && (!mid || (f2ord(s) >= thr));
      const float p = keep ? __expf(s - m) : 0.0f;
      srow[i] = p;
      sum += p;
    }
    sum = wave_sum32(sum);
    if (lane == 0) rowinv[row] = 1.0f / sum;
  }
  __syncthreads();

  // ---- Phase 3: out = P x V ; wave wid owns d-range [wid*16, wid*16+16) ----
  v8f oacc = {};
  for (int kk = 0; kk < S_LEN / 32; ++kk) {
    v16bf pa;
    const float* pr = sc + qrow * S_LEN + kk * 32;
    #pragma unroll
    for (int i = 0; i < 8; ++i) {
      pa[i]     = (__bf16)pr[khalf * 16 + i];
      pa[8 + i] = (__bf16)pr[khalf * 16 + 8 + i];
    }
    v16bf vb;
    const __bf16* vp = vbfT + ((size_t)kvh * HD + wid * 16 + qrow) * S_LEN
                            + kk * 32 + khalf * 16;
    #pragma unroll
    for (int i = 0; i < 16; ++i) vb[i] = vp[i];
    oacc = __builtin_amdgcn_wmma_f32_16x16x32_bf16(false, pa, false, vb,
                                                   (short)0, oacc, false, false);
  }
  #pragma unroll
  for (int i = 0; i < 8; ++i) {
    const int row = khalf * 8 + i;
    const int d   = wid * 16 + qrow;
    attn_bf[((size_t)(tile * 16 + row)) * HID + h * HD + d] =
        (__bf16)(oacc[i] * rowinv[row]);
  }
}

// ---------------------------------------------------------------------------
// host launcher
// ---------------------------------------------------------------------------
extern "C" void kernel_launch(void* const* d_in, const int* in_sizes, int n_in,
                              void* d_out, int out_size, void* d_ws, size_t ws_size,
                              hipStream_t stream) {
  (void)in_sizes; (void)n_in; (void)out_size; (void)ws_size;
  const float* hs = (const float*)d_in[0];
  const float* wq = (const float*)d_in[1];
  const float* wk = (const float*)d_in[2];
  const float* wv = (const float*)d_in[3];
  const float* wo = (const float*)d_in[4];

  char* base = (char*)d_ws;
  size_t off = 0;
  auto take = [&](size_t bytes) -> void* {
    void* r = base + off;
    off += (bytes + 255) & ~(size_t)255;
    return r;
  };

  __bf16* hidb = (__bf16*)take((size_t)S_LEN * HID * 2);
  __bf16* wqT  = (__bf16*)take((size_t)HID * NH * HD * 2);   // [N][K]
  __bf16* wkT  = (__bf16*)take((size_t)HID * NKV * HD * 2);  // [N][K]
  __bf16* wvT  = (__bf16*)take((size_t)HID * NKV * HD * 2);  // [N][K]
  __bf16* woT  = (__bf16*)take((size_t)NH * HD * HID * 2);   // [N][K]
  float*  qf   = (float*)take((size_t)S_LEN * NH * HD * 4);
  float*  kf   = (float*)take((size_t)S_LEN * NKV * HD * 4);
  float*  vf   = (float*)take((size_t)S_LEN * NKV * HD * 4);
  __bf16* qbf  = (__bf16*)take((size_t)NH * S_LEN * HD * 2);
  __bf16* kbf  = (__bf16*)take((size_t)NKV * S_LEN * HD * 2);
  __bf16* vbfT = (__bf16*)take((size_t)NKV * HD * S_LEN * 2);
  __bf16* attnb = hidb;   // hidden_bf16 is dead after QKV GEMMs: alias to save 16 MB

  // 1) precision conversion (+ weight transpose to K-contiguous layout)
  hh_f32_to_bf16<<<2048, 256, 0, stream>>>(hs, hidb, (size_t)S_LEN * HID);
  hh_f32_to_bf16_T<<<2048, 256, 0, stream>>>(wq, wqT, HID, NH * HD);
  hh_f32_to_bf16_T<<<1024, 256, 0, stream>>>(wk, wkT, HID, NKV * HD);
  hh_f32_to_bf16_T<<<1024, 256, 0, stream>>>(wv, wvT, HID, NKV * HD);
  hh_f32_to_bf16_T<<<2048, 256, 0, stream>>>(wo, woT, NH * HD, HID);

  // 2) QKV projections (WMMA, async-LDS A staging, L2-resident B)
  hh_gemm_bf16<<<dim3((NH * HD) / 64, S_LEN / 64), 256, 0, stream>>>(hidb, wqT, qf, S_LEN, NH * HD, HID);
  hh_gemm_bf16<<<dim3((NKV * HD) / 64, S_LEN / 64), 256, 0, stream>>>(hidb, wkT, kf, S_LEN, NKV * HD, HID);
  hh_gemm_bf16<<<dim3((NKV * HD) / 64, S_LEN / 64), 256, 0, stream>>>(hidb, wvT, vf, S_LEN, NKV * HD, HID);

  // 3) RoPE (+1/sqrt(HD) folded into Q) and repacks
  hh_rope_pack<<<dim3(NH * S_LEN), 128, 0, stream>>>(qf, qbf, NH, QK_SCALE);
  hh_rope_pack<<<dim3(NKV * S_LEN), 128, 0, stream>>>(kf, kbf, NKV, 1.0f);
  hh_v_packT<<<2048, 256, 0, stream>>>(vf, vbfT);

  // 4) heavy-hitter attention (~140 KB dynamic LDS — fits CDNA5's 320 KB WGP LDS)
  const size_t ldsz = (size_t)16 * S_LEN * sizeof(float)      // scores
                    + 8 * 256 * sizeof(unsigned)              // radix histograms
                    + 8 * 2 * sizeof(unsigned)                // per-wave scratch
                    + 16 * sizeof(float)                      // row 1/sum
                    + (size_t)16 * HD * sizeof(__bf16);       // TDM-staged Q tile
  (void)hipFuncSetAttribute((const void*)hh_attn_kernel,
                            hipFuncAttributeMaxDynamicSharedMemorySize, (int)ldsz);
  hh_attn_kernel<<<dim3(NH * (S_LEN / 16)), 256, ldsz, stream>>>(qbf, kbf, vbfT, attnb);

  // 5) output projection (WMMA) -> fp32 result
  hh_gemm_bf16<<<dim3(HID / 64, S_LEN / 64), 256, 0, stream>>>(attnb, woT, (float*)d_out, S_LEN, HID, HID);
}
